// Attention_65317862638379
// MI455X (gfx1250) — matrix operand order repaired and checked by reading, not compiled, not run on hardware
//
#include <hip/hip_runtime.h>
#include <hip/hip_bf16.h>

// MI455X / gfx1250: wave32, WMMA 16x16x32 bf16 with fp32 accumulate.
// Pipeline: qkv GEMM+rotary -> flash block-attention -> inverse-rotary -> out GEMM.
// Workspace: q,k,v in (b,h,512,256) bf16 (64MB each) + attn-out (B*N,1024) bf16 (64MB).

typedef __bf16 bf16;
typedef __attribute__((ext_vector_type(16))) __bf16 v16bf;
typedef __attribute__((ext_vector_type(8)))  float  v8f;

union BF16x16 { uint4 u[2]; v16bf v; };

#define WMMA_BF16(a, b, c) \
  __builtin_amdgcn_wmma_f32_16x16x32_bf16(false, (a), false, (b), (short)0, (c), false, false)

// Problem constants
#define NB   16
#define NN   2048
#define NDIM 1024
#define NH   16
#define NDH  64
// Block-attention: 512 rows x 256 features per (b,h)

__device__ __forceinline__ void wait_asynccnt0() {
#if __has_builtin(__builtin_amdgcn_s_wait_asynccnt)
  __builtin_amdgcn_s_wait_asynccnt(0);
#else
  asm volatile("s_wait_asynccnt 0" ::: "memory");
#endif
}

// ---------------------------------------------------------------------------
// Kernel A: qkv = x @ W_qkv^T  (M=B*N=32768, N=3072, K=1024), bf16 WMMA,
// fused rotary (pairs are adjacent cols -> adjacent lanes in C layout) and
// v trans-add; outputs q/k/v in block layout [(b*H+h)*512 + j]*256 + d.
// ---------------------------------------------------------------------------
__global__ __launch_bounds__(128)
void qkv_rot_kernel(const float* __restrict__ x, const float* __restrict__ Wq,
                    const float* __restrict__ angles, const float* __restrict__ trans,
                    const float* __restrict__ tcoef,
                    bf16* __restrict__ qw, bf16* __restrict__ kw, bf16* __restrict__ vw)
{
  const int lane = threadIdx.x & 31;
  const int wave = threadIdx.x >> 5;
  const int half = lane >> 4;          // wave32 halves (WMMA lane split)
  const int l15  = lane & 15;
  const int tm   = blockIdx.x;             // 16-row tile of B*N
  const int tn   = blockIdx.y * 4 + wave;  // 16-col tile of 3072

  const float* xrow = x  + (size_t)(tm * 16 + l15) * NDIM;   // A row (M = lane)
  const float* wrow = Wq + (size_t)(tn * 16 + l15) * NDIM;   // B col (N = lane), K contiguous

  v8f acc = {};
  for (int k0 = 0; k0 < NDIM; k0 += 32) {
    // A fragment: 16x32 bf16, lane holds K = ka..ka+15
    const int ka = k0 + (half ? 16 : 0);
    float af[16];
    *(float4*)&af[0]  = *(const float4*)&xrow[ka];
    *(float4*)&af[4]  = *(const float4*)&xrow[ka + 4];
    *(float4*)&af[8]  = *(const float4*)&xrow[ka + 8];
    *(float4*)&af[12] = *(const float4*)&xrow[ka + 12];
    v16bf a = {};
#pragma unroll
    for (int t = 0; t < 16; ++t) a[t] = (bf16)af[t];

    // B fragment: 32x16 bf16, lane half selects K = kb..kb+15 (contiguous in W row)
    const int kb = k0 + (half ? 16 : 0);
    float bfv[16];
#pragma unroll
    for (int t = 0; t < 4; ++t) *(float4*)&bfv[t * 4] = *(const float4*)&wrow[kb + t * 4];
    v16bf bb = {};
#pragma unroll
    for (int t = 0; t < 16; ++t) bb[t] = (bf16)bfv[t];

    if (k0 + 32 < NDIM) {
      __builtin_prefetch(&wrow[kb + 32], 0, 1);   // global_prefetch_b8
      __builtin_prefetch(&xrow[ka + 32], 0, 1);
    }
    acc = WMMA_BF16(a, bb, acc);
  }

  // Epilogue: C layout element (r, lane) = (M = r + 8*half, N = l15).
  const int which = tn >> 6;                 // 0=q 1=k 2=v
  const int cc    = (tn & 63) * 16 + l15;    // column within 1024
  const int h  = cc >> 6;
  const int dh = cc & 63;
  bf16* dst = (which == 0) ? qw : (which == 1) ? kw : vw;
  const float c  = tcoef[0];
  const bool rot = (dh >= 32);               // wave-uniform (16-aligned window)
#pragma unroll
  for (int r = 0; r < 8; ++r) {
    const int row = tm * 16 + r + half * 8;
    const int b = row >> 11, n = row & 2047;
    float val = acc[r];
    if (rot) {
      const int p = (dh - 32) >> 1;
      const float ang = angles[((size_t)b * NN + n) * 16 + p];
      const float cs = __cosf(ang), sn = __sinf(ang);
      const float partner = __shfl_xor(val, 1);  // pair lives in adjacent lane
      val = (dh & 1) ? (partner * sn + val * cs) : (val * cs - partner * sn);
      if (which == 2) val += c * trans[((size_t)b * NN + n) * 32 + (dh - 32)];
    }
    const int j = n >> 2, blk = n & 3;
    const int d = blk * 64 + dh;
    dst[(((size_t)(b * NH + h)) * 512 + j) * 256 + d] = (bf16)val;
  }
}

// ---------------------------------------------------------------------------
// Kernel B: flash block-attention per (b,h). 8 waves x 16 query block-rows.
// K staged to LDS via async global->LDS copies (ASYNCcnt), V staged transposed,
// P round-trips through LDS for the C->A fragment transpose.
// Fused -c*t and inverse rotary epilogue.
// ---------------------------------------------------------------------------
__global__ __launch_bounds__(256)
void attn_kernel(const bf16* __restrict__ qw, const bf16* __restrict__ kw,
                 const bf16* __restrict__ vw, const float* __restrict__ angles,
                 const float* __restrict__ trans, const float* __restrict__ tcoef,
                 bf16* __restrict__ ao)
{
  __shared__ __align__(32) bf16  k_lds[32 * 256];     // [key][d]      16 KB
  __shared__ __align__(32) bf16  v_lds[256 * 32];     // [d][key]      16 KB
  __shared__ __align__(16) float p_lds[8][16 * 32];   // per-wave P    16 KB

  const int tid = threadIdx.x, lane = tid & 31, wave = tid >> 5;
  const int half = lane >> 4, l15 = lane & 15;
  const int bh = blockIdx.x;                          // b*H + h
  const int rowBase = blockIdx.y * 128 + wave * 16;   // block-row base (0..511)
  const int b = bh >> 4, h = bh & 15;
  const float c  = tcoef[0];
  const float SC = 0.0625f;                           // (DH*BLK)^-0.5 / TAU

  // Q fragments (16 rows x 256 features, 8 K-chunks of 32)
  const bf16* qrp = qw + ((size_t)bh * 512 + rowBase + l15) * 256;
  v16bf qf[8];
#pragma unroll
  for (int f = 0; f < 8; ++f) {
    const int kg = f * 32 + (half ? 16 : 0);
    BF16x16 u;
    u.u[0] = *(const uint4*)&qrp[kg];
    u.u[1] = *(const uint4*)&qrp[kg + 8];
    qf[f] = u.v;
  }

  const v8f vzero = {};
  v8f o[16];
#pragma unroll
  for (int dt = 0; dt < 16; ++dt) o[dt] = vzero;
  float mrun[8], srun[8];
#pragma unroll
  for (int r = 0; r < 8; ++r) { mrun[r] = -1e30f; srun[r] = 0.f; }

  for (int it = 0; it < 16; ++it) {
    const int jb = it * 32;
    // --- Cooperative staging ---
    // K: linear 16B/lane global->LDS copies via the async path (ASYNCcnt).
#pragma unroll
    for (int i = 0; i < 4; ++i) {
      const int off = (tid + i * 256) * 8;            // linear bf16 offset in 32x256 block
      const bf16* gsrc = &kw[((size_t)bh * 512 + jb) * 256 + off];
      const unsigned ldst = (unsigned)(uintptr_t)&k_lds[off];  // ISA: LDS addr = addr[31:0]
      asm volatile("global_load_async_to_lds_b128 %0, %1, off"
                   :: "v"(ldst), "v"(gsrc) : "memory");
    }
    // V: transposed scatter (regular load + ds stores).
#pragma unroll
    for (int i = 0; i < 4; ++i) {
      const int off = (tid + i * 256) * 8;
      const int kk = off >> 8, d0 = off & 255;
      const uint4 raw = *(const uint4*)&vw[((size_t)bh * 512 + jb + kk) * 256 + d0];
      const bf16* e = (const bf16*)&raw;
#pragma unroll
      for (int t = 0; t < 8; ++t) v_lds[(d0 + t) * 32 + kk] = e[t];
    }
    wait_asynccnt0();
    __syncthreads();

    // S = Q K^T : two 16-key tiles
    v8f s0 = {}, s1 = {};
#pragma unroll
    for (int f = 0; f < 8; ++f) {
      const int db = f * 32 + (half ? 16 : 0);
      const v16bf b0 = *(const v16bf*)&k_lds[l15 * 256 + db];
      const v16bf b1 = *(const v16bf*)&k_lds[(l15 + 16) * 256 + db];
      s0 = WMMA_BF16(qf[f], b0, s0);
      s1 = WMMA_BF16(qf[f], b1, s1);
    }

    // Online softmax (row = r + 8*half; cols across the 16-lane half)
    float corr[8];
#pragma unroll
    for (int r = 0; r < 8; ++r) {
      const float a0 = s0[r] * SC, a1 = s1[r] * SC;
      float mx = fmaxf(a0, a1);
#pragma unroll
      for (int m = 1; m < 16; m <<= 1) mx = fmaxf(mx, __shfl_xor(mx, m));
      const float mn = fmaxf(mrun[r], mx);
      const float cr = __expf(mrun[r] - mn);
      const float p0 = __expf(a0 - mn), p1 = __expf(a1 - mn);
      float ls = p0 + p1;
#pragma unroll
      for (int m = 1; m < 16; m <<= 1) ls += __shfl_xor(ls, m);
      srun[r] = srun[r] * cr + ls;
      mrun[r] = mn;
      corr[r] = cr;
      const int row = r + half * 8;
      p_lds[wave][row * 32 + l15]      = p0;
      p_lds[wave][row * 32 + 16 + l15] = p1;
    }
#pragma unroll
    for (int dt = 0; dt < 16; ++dt)
#pragma unroll
      for (int r = 0; r < 8; ++r) o[dt][r] *= corr[r];

    // P (16x32) as A fragment via per-wave LDS (same-wave DS ops are in order)
    const float* pr = &p_lds[wave][l15 * 32];
    const int kg = half ? 16 : 0;
    v16bf pf = {};
#pragma unroll
    for (int t = 0; t < 8; ++t) {
      pf[t]     = (bf16)pr[kg + t];
      pf[t + 8] = (bf16)pr[kg + 8 + t];
    }

    // O += P @ V
#pragma unroll
    for (int dt = 0; dt < 16; ++dt) {
      const int d = dt * 16 + l15;
      const v16bf vf = *(const v16bf*)&v_lds[d * 32 + half * 16];
      o[dt] = WMMA_BF16(pf, vf, o[dt]);
    }
    __syncthreads();
  }

  // Epilogue: normalize, -c*t, inverse rotary, write (b,n,h*64+dh) bf16.
  float inv[8];
#pragma unroll
  for (int r = 0; r < 8; ++r) inv[r] = 1.0f / srun[r];
#pragma unroll
  for (int dt = 0; dt < 16; ++dt) {
    const int d = dt * 16 + l15;
    const int blk = d >> 6, dh = d & 63;
    const bool rot = (dh >= 32);                     // wave-uniform per dt
#pragma unroll
    for (int r = 0; r < 8; ++r) {
      const int j = rowBase + r + half * 8;
      const int n = j * 4 + blk;
      float val = o[dt][r] * inv[r];
      if (rot) {
        val -= c * trans[((size_t)b * NN + n) * 32 + (dh - 32)];
        const float partner = __shfl_xor(val, 1);
        const int p = (dh - 32) >> 1;
        const float ang = angles[((size_t)b * NN + n) * 16 + p];
        const float cs = __cosf(ang), sn = __sinf(ang);
        // _apply_rot with (cos, -sin)
        val = (dh & 1) ? (val * cs - partner * sn) : (val * cs + partner * sn);
      }
      ao[((size_t)b * NN + n) * 1024 + h * 64 + dh] = (bf16)val;
    }
  }
}

// ---------------------------------------------------------------------------
// Kernel C: out = ao @ W_out^T + b_out  (32768 x 1024, K=1024), fp32 store.
// ---------------------------------------------------------------------------
__global__ __launch_bounds__(128)
void out_gemm_kernel(const bf16* __restrict__ ao, const float* __restrict__ Wo,
                     const float* __restrict__ bo, float* __restrict__ out)
{
  const int lane = threadIdx.x & 31;
  const int wave = threadIdx.x >> 5;
  const int half = lane >> 4;
  const int l15  = lane & 15;
  const int tm   = blockIdx.x;
  const int tn   = blockIdx.y * 4 + wave;   // 0..63

  const bf16*  arp  = ao + (size_t)(tm * 16 + l15) * 1024;
  const float* wrow = Wo + (size_t)(tn * 16 + l15) * 1024;

  v8f acc = {};
  for (int k0 = 0; k0 < 1024; k0 += 32) {
    const int ka = k0 + (half ? 16 : 0);
    BF16x16 ua;
    ua.u[0] = *(const uint4*)&arp[ka];
    ua.u[1] = *(const uint4*)&arp[ka + 8];

    const int kb = k0 + (half ? 16 : 0);
    float bfv[16];
#pragma unroll
    for (int t = 0; t < 4; ++t) *(float4*)&bfv[t * 4] = *(const float4*)&wrow[kb + t * 4];
    v16bf bb = {};
#pragma unroll
    for (int t = 0; t < 16; ++t) bb[t] = (bf16)bfv[t];

    if (k0 + 32 < 1024) __builtin_prefetch(&wrow[kb + 32], 0, 1);
    acc = WMMA_BF16(ua.v, bb, acc);
  }

  const int ncol = tn * 16 + l15;
  const float bias = bo[ncol];
#pragma unroll
  for (int r = 0; r < 8; ++r) {
    const int row = tm * 16 + r + half * 8;
    out[(size_t)row * 1024 + ncol] = acc[r] + bias;
  }
}

// ---------------------------------------------------------------------------
extern "C" void kernel_launch(void* const* d_in, const int* in_sizes, int n_in,
                              void* d_out, int out_size, void* d_ws, size_t ws_size,
                              hipStream_t stream)
{
  (void)in_sizes; (void)n_in; (void)out_size; (void)ws_size;
  const float* x      = (const float*)d_in[0];
  const float* angles = (const float*)d_in[1];
  const float* trans  = (const float*)d_in[2];
  const float* Wq     = (const float*)d_in[3];
  const float* Wo     = (const float*)d_in[4];
  const float* bo     = (const float*)d_in[5];
  const float* tc     = (const float*)d_in[6];
  float* out = (float*)d_out;

  char* ws = (char*)d_ws;
  const size_t SEG = (size_t)NB * NH * 512 * 256 * sizeof(bf16);  // 64 MB
  bf16* qw = (bf16*)(ws);
  bf16* kw = (bf16*)(ws + SEG);
  bf16* vw = (bf16*)(ws + 2 * SEG);
  bf16* ao = (bf16*)(ws + 3 * SEG);

  qkv_rot_kernel<<<dim3(2048, 48), 128, 0, stream>>>(x, Wq, angles, trans, tc, qw, kw, vw);
  attn_kernel  <<<dim3(256, 4),   256, 0, stream>>>(qw, kw, vw, angles, trans, tc, ao);
  out_gemm_kernel<<<dim3(2048, 16), 128, 0, stream>>>(ao, Wo, bo, out);
}
